// GraphSageModel_54863912239933
// MI455X (gfx1250) — matrix operand order repaired and checked by reading, not compiled, not run on hardware
//
#include <hip/hip_runtime.h>

// ---------------------------------------------------------------------------
// GraphSAGE (2x SAGEConv + FC) for MI455X (gfx1250, wave32, WMMA bf16)
//   - persistent-block GEMM: B fragments live in registers, reused across
//     M-tiles (cuts weight L2 traffic ~5x)
//   - global_prefetch for the streaming A operand
//   - 32-bit index math in the hot path (no v_mul_u64 in the epilogue)
//   - 128-bit vectorized streaming passes, native f32 atomics for scatter
// ---------------------------------------------------------------------------

typedef __attribute__((ext_vector_type(16))) __bf16 v16bf;
typedef __attribute__((ext_vector_type(8)))  __bf16 v8bf;
typedef __attribute__((ext_vector_type(4)))  __bf16 v4bf;
typedef __attribute__((ext_vector_type(8)))  float  v8f;

#define GS_NODES 50000
#define GS_EDGES 800000
#define GS_D     128      // feature width of x / hidden layers
#define GS_DOUT  64

// ---------------------------------------------------------------- utilities
__global__ void gs_fill_zero4(float4* __restrict__ p, int n4) {
  int i = blockIdx.x * blockDim.x + threadIdx.x;
  if (i < n4) p[i] = make_float4(0.f, 0.f, 0.f, 0.f);
}

__global__ void gs_f32_to_bf16_4(const float4* __restrict__ s,
                                 v4bf* __restrict__ d, int n4) {
  int i = blockIdx.x * blockDim.x + threadIdx.x;
  if (i < n4) {
    float4 v = s[i];
    v4bf o; o[0] = (__bf16)v.x; o[1] = (__bf16)v.y;
            o[2] = (__bf16)v.z; o[3] = (__bf16)v.w;
    d[i] = o;
  }
}

// scale aggregated sums by 1/deg and narrow to bf16 (4 feats/thread; the 4
// features always belong to the same node since 4 | 128)
__global__ void gs_scale_bf16_4(const float4* __restrict__ agg,
                                const float* __restrict__ dinv,
                                v4bf* __restrict__ out, int n4) {
  int i = blockIdx.x * blockDim.x + threadIdx.x;
  if (i < n4) {
    float  s = dinv[(i << 2) >> 7];
    float4 v = agg[i];
    v4bf o; o[0] = (__bf16)(v.x * s); o[1] = (__bf16)(v.y * s);
            o[2] = (__bf16)(v.z * s); o[3] = (__bf16)(v.w * s);
    out[i] = o;
  }
}

// ----------------------------------------------------------------- degrees
__global__ void gs_deg_count(const long long* __restrict__ dst,
                             float* __restrict__ deg) {
  int e = blockIdx.x * blockDim.x + threadIdx.x;
  if (e < GS_EDGES) unsafeAtomicAdd(&deg[(int)dst[e]], 1.0f);
}

__global__ void gs_deg_invert(float* __restrict__ deg) {
  int i = blockIdx.x * blockDim.x + threadIdx.x;
  if (i < GS_NODES) deg[i] = 1.0f / fmaxf(deg[i], 1.0f);
}

// -------------------------------------------------------- mean aggregation
// one wave per edge; each lane moves 4 contiguous floats (float4) and
// scatter-adds them with native f32 atomics (L2-resident agg buffer).
__global__ void gs_aggregate(const long long* __restrict__ src,
                             const long long* __restrict__ dst,
                             const float* __restrict__ x,
                             float* __restrict__ agg) {
  int e    = blockIdx.x * (blockDim.x >> 5) + (threadIdx.x >> 5);
  int lane = threadIdx.x & 31;
  if (e >= GS_EDGES) return;
  int s = (int)src[e];
  int d = (int)dst[e];
  const float4 v = *(const float4*)(x + s * GS_D + lane * 4);
  float* o = agg + d * GS_D + lane * 4;
  unsafeAtomicAdd(o + 0, v.x);
  unsafeAtomicAdd(o + 1, v.y);
  unsafeAtomicAdd(o + 2, v.z);
  unsafeAtomicAdd(o + 3, v.w);
}

// ------------------------------------------------- weight pack (B fragment)
// Pack W[K][N] (row major, f32) into the per-lane layout consumed by
// V_WMMA_F32_16X16X32_BF16's B operand:
//   b[e] = B[k = (e>>3)*16 + (lane>>4)*8 + (e&7)][n = lane&15]  within a 32x16 (K x N) tile
// Packed index: ((kt*(N/16) + nt)*32 + lane)*16 + e  -> contiguous 32B/lane.
__global__ void gs_pack_w(const float* __restrict__ W,
                          __bf16* __restrict__ P, int K, int N) {
  int i = blockIdx.x * blockDim.x + threadIdx.x;
  if (i >= K * N) return;
  int e      = i & 15;
  int lane   = (i >> 4) & 31;
  int tile   = i >> 9;
  int nTiles = N >> 4;
  int nt = tile % nTiles;
  int kt = tile / nTiles;
  int n = nt * 16 + (lane & 15);
  int k = kt * 32 + (e >> 3) * 16 + (lane >> 4) * 8 + (e & 7);
  P[i] = (__bf16)W[k * N + n];
}

// --------------------------------------------------------- fused SAGE GEMM
// out[m,n] = act( sum_k A1[m,k]*B1[k,n] (+ A2[m,k]*B2[k,n]) + bias[n] )
// Persistent blocks: each block owns nTiles N-tiles (1 wave each), loads all
// B fragments into registers ONCE, then streams M-tiles. A row-major bf16.
// All hot-path indices fit in 32 bits (max 6.4M) -> int addressing only.
template <int KT, bool HAS2, bool RELU>
__global__ void gs_gemm_wmma(const __bf16* __restrict__ A1,
                             const __bf16* __restrict__ B1,
                             const __bf16* __restrict__ A2,
                             const __bf16* __restrict__ B2,
                             const float* __restrict__ bias,
                             float* __restrict__ out,
                             int N, int K, int mTiles) {
  const int lane   = threadIdx.x & 31;
  const int nt     = threadIdx.x >> 5;     // wave id == N-tile within block
  const int nTiles = blockDim.x >> 5;
  const int row    = lane & 15;            // A row / C column selector
  const int half   = lane >> 4;

  // ---- B fragments: load once, keep in VGPRs for all M-tiles
  v16bf b1r[KT], b2r[KT];
  #pragma unroll
  for (int kt = 0; kt < KT; ++kt) {
    b1r[kt] = *(const v16bf*)(B1 + ((kt * nTiles + nt) * 32 + lane) * 16);
    if (HAS2)
      b2r[kt] = *(const v16bf*)(B2 + ((kt * nTiles + nt) * 32 + lane) * 16);
  }
  const float bv = bias[nt * 16 + row];

  for (int mt = blockIdx.x; mt < mTiles; mt += gridDim.x) {
    const int arow = (mt * 16 + row) * K;

    // prefetch the next M-tile's A rows (global_prefetch path)
    const int mn = mt + gridDim.x;
    if (mn < mTiles) {
      const int prow = (mn * 16 + row) * K;
      __builtin_prefetch(A1 + prow + half * 32, 0, 0);
      if (HAS2) __builtin_prefetch(A2 + prow + half * 32, 0, 0);
    }

    v8f acc = {0.f, 0.f, 0.f, 0.f, 0.f, 0.f, 0.f, 0.f};
    #pragma unroll
    for (int kt = 0; kt < KT; ++kt) {
      const int kb = kt * 32 + half * 8;
      v8bf lo = *(const v8bf*)(A1 + arow + kb);
      v8bf hi = *(const v8bf*)(A1 + arow + kb + 16);
      v16bf a;
      #pragma unroll
      for (int i = 0; i < 8; ++i) { a[i] = lo[i]; a[i + 8] = hi[i]; }
      acc = __builtin_amdgcn_wmma_f32_16x16x32_bf16(
          false, a, false, b1r[kt], (short)0, acc, false, false);
    }
    if (HAS2) {
      #pragma unroll
      for (int kt = 0; kt < KT; ++kt) {
        const int kb = kt * 32 + half * 8;
        v8bf lo = *(const v8bf*)(A2 + arow + kb);
        v8bf hi = *(const v8bf*)(A2 + arow + kb + 16);
        v16bf a;
        #pragma unroll
        for (int i = 0; i < 8; ++i) { a[i] = lo[i]; a[i + 8] = hi[i]; }
        acc = __builtin_amdgcn_wmma_f32_16x16x32_bf16(
            false, a, false, b2r[kt], (short)0, acc, false, false);
      }
    }

    // epilogue: one base address, constant row displacements (v * N)
    float* __restrict__ p = out + (mt * 16 + half * 8) * N + nt * 16 + row;
    #pragma unroll
    for (int v = 0; v < 8; ++v) {
      float r = acc[v] + bv;
      if (RELU) r = fmaxf(r, 0.0f);
      p[v * N] = r;
    }
  }
}

// ---------------------------------------------------------------------------
extern "C" void kernel_launch(void* const* d_in, const int* in_sizes, int n_in,
                              void* d_out, int out_size, void* d_ws, size_t ws_size,
                              hipStream_t stream) {
  (void)in_sizes; (void)n_in; (void)out_size; (void)ws_size;

  // inputs (setup_inputs order)
  const float*     x    = (const float*)d_in[0];
  const long long* ei   = (const long long*)d_in[1];   // int64 [2, E]
  const float*     Wl0  = (const float*)d_in[2];
  const float*     Wr0  = (const float*)d_in[3];
  const float*     b0   = (const float*)d_in[4];
  const float*     Wl1  = (const float*)d_in[5];
  const float*     Wr1  = (const float*)d_in[6];
  const float*     b1   = (const float*)d_in[7];
  const float*     Wfc  = (const float*)d_in[8];
  const float*     bfc  = (const float*)d_in[9];
  const long long* srcI = ei;
  const long long* dstI = ei + GS_EDGES;

  // workspace carve-up (all offsets 256B-aligned)
  const long NF  = (long)GS_NODES * GS_D;              // 6,400,000
  const int  NF4 = (int)(NF >> 2);
  char* ws = (char*)d_ws;
  float*  deg   = (float*)ws;                 ws += 50176 * sizeof(float);
  float*  aggF  = (float*)ws;                 ws += NF * sizeof(float);
  __bf16* inBF  = (__bf16*)ws;                ws += NF * sizeof(__bf16);
  __bf16* aggBF = (__bf16*)ws;                ws += NF * sizeof(__bf16);
  float*  h0F   = (float*)ws;                 ws += NF * sizeof(float);
  float*  h1F   = (float*)ws;                 ws += NF * sizeof(float);
  __bf16* pWl0  = (__bf16*)ws;                ws += 16384 * sizeof(__bf16);
  __bf16* pWr0  = (__bf16*)ws;                ws += 16384 * sizeof(__bf16);
  __bf16* pWl1  = (__bf16*)ws;                ws += 16384 * sizeof(__bf16);
  __bf16* pWr1  = (__bf16*)ws;                ws += 16384 * sizeof(__bf16);
  __bf16* pWfc  = (__bf16*)ws;                ws += 8192  * sizeof(__bf16);

  const int T = 256;
  #define GB(n) ((int)(((long)(n) + T - 1) / T))

  // degrees
  gs_fill_zero4<<<GB(50176 / 4), T, 0, stream>>>((float4*)deg, 50176 / 4);
  gs_deg_count <<<GB(GS_EDGES), T, 0, stream>>>(dstI, deg);
  gs_deg_invert<<<GB(GS_NODES), T, 0, stream>>>(deg);

  // pack weights into WMMA B-fragment layout (bf16)
  gs_pack_w<<<GB(16384), T, 0, stream>>>(Wl0, pWl0, GS_D, GS_D);
  gs_pack_w<<<GB(16384), T, 0, stream>>>(Wr0, pWr0, GS_D, GS_D);
  gs_pack_w<<<GB(16384), T, 0, stream>>>(Wl1, pWl1, GS_D, GS_D);
  gs_pack_w<<<GB(16384), T, 0, stream>>>(Wr1, pWr1, GS_D, GS_D);
  gs_pack_w<<<GB(8192),  T, 0, stream>>>(Wfc, pWfc, GS_D, GS_DOUT);

  const int mTiles   = GS_NODES / 16;                  // 3125 (exact)
  const int gemmGrid = 625;                            // 5 M-tiles / block
  const int aggGrid  = GB((long)GS_EDGES * 32);        // 1 wave / edge

  // ---- layer 0: h0 = relu(mean_agg(x)@Wl0 + x@Wr0 + b0)
  gs_fill_zero4  <<<GB(NF4), T, 0, stream>>>((float4*)aggF, NF4);
  gs_f32_to_bf16_4<<<GB(NF4), T, 0, stream>>>((const float4*)x, (v4bf*)inBF, NF4);
  gs_aggregate   <<<aggGrid, T, 0, stream>>>(srcI, dstI, x, aggF);
  gs_scale_bf16_4<<<GB(NF4), T, 0, stream>>>((const float4*)aggF, deg,
                                             (v4bf*)aggBF, NF4);
  gs_gemm_wmma<4, true, true><<<gemmGrid, 256, 0, stream>>>(
      aggBF, pWl0, inBF, pWr0, b0, h0F, GS_D, GS_D, mTiles);

  // ---- layer 1: h1 = relu(mean_agg(h0)@Wl1 + h0@Wr1 + b1)
  gs_fill_zero4  <<<GB(NF4), T, 0, stream>>>((float4*)aggF, NF4);
  gs_f32_to_bf16_4<<<GB(NF4), T, 0, stream>>>((const float4*)h0F, (v4bf*)inBF, NF4);
  gs_aggregate   <<<aggGrid, T, 0, stream>>>(srcI, dstI, h0F, aggF);
  gs_scale_bf16_4<<<GB(NF4), T, 0, stream>>>((const float4*)aggF, deg,
                                             (v4bf*)aggBF, NF4);
  gs_gemm_wmma<4, true, true><<<gemmGrid, 256, 0, stream>>>(
      aggBF, pWl1, inBF, pWr1, b1, h1F, GS_D, GS_D, mTiles);

  // ---- fc: out = h1 @ Wfc + bfc   (N=64 -> 4 waves/block, no relu)
  gs_f32_to_bf16_4<<<GB(NF4), T, 0, stream>>>((const float4*)h1F, (v4bf*)inBF, NF4);
  gs_gemm_wmma<4, false, false><<<gemmGrid, 128, 0, stream>>>(
      inBF, pWfc, nullptr, nullptr, bfc, (float*)d_out, GS_DOUT, GS_D, mTiles);
  #undef GB
}
